// Convolution_56599079027372
// MI455X (gfx1250) — matrix operand order repaired and checked by reading, not compiled, not run on hardware
//
#include <hip/hip_runtime.h>

#define IN_C   256
#define IMG    224
#define KS     7
#define OUTW   218                       // 224 - 7 + 1
#define TILE_M 16                        // ow per wave
#define NWAVE  4
#define TILE_W (TILE_M * NWAVE)          // 64 ow per workgroup
#define TILE_H 16                        // oh rows per workgroup
#define POSW   (TILE_W + KS - 1)         // 70 staged input columns
#define NSTEP  (KS * IN_C / 32)          // 56 K-steps of 32 (k = kw*256 + c)

// ---- LDS layout (bytes) ----
#define B_HI_OFF   0
#define B_BYTES    (NSTEP * 1024)                  // 57344: 56 steps x (32x16 bf16)
#define B_LO_OFF   (B_HI_OFF + B_BYTES)
#define XT_HI_OFF  (B_LO_OFF + B_BYTES)            // 114688
#define XT_BYTES   (POSW * IN_C * 2)               // 35840
#define XT_LO_OFF  (XT_HI_OFF + XT_BYTES)
#define OUT_OFF    (XT_LO_OFF + XT_BYTES)          // 186368
#define OUT_BYTES  (NWAVE * TILE_H * TILE_M * 4)   // 4096
#define SMEM_BYTES (OUT_OFF + OUT_BYTES)           // 190464 (< 320KB WGP LDS)

typedef __attribute__((ext_vector_type(16))) __bf16 v16bf;
typedef __attribute__((ext_vector_type(8)))  float  v8f;

union Frag16 { uint4 q[2]; v16bf v; };

static __device__ __forceinline__ unsigned short bf16_rn(float f) {
  unsigned int u = __float_as_uint(f);
  unsigned int r = u + 0x7FFFu + ((u >> 16) & 1u);   // round-to-nearest-even
  return (unsigned short)(r >> 16);
}
static __device__ __forceinline__ float bf16_tof(unsigned short h) {
  return __uint_as_float(((unsigned int)h) << 16);
}

__global__ __launch_bounds__(128) void conv7x7_wmma_bf16x3(
    const float* __restrict__ x, const float* __restrict__ w,
    const float* __restrict__ bias, float* __restrict__ out)
{
  __shared__ __align__(16) unsigned char smem[SMEM_BYTES];
  const int tid  = threadIdx.x;
  const int lane = tid & 31;
  const int wv   = tid >> 5;
  const int b    = blockIdx.x;
  const int oh0  = blockIdx.y * TILE_H;
  const int ow0  = blockIdx.z * TILE_W;

  // ---- phase 0: weights -> LDS in exact 32x16 bf16 B-fragment layout (hi & lo planes)
  // step s: kw = s>>3, c0 = (s&7)*32. Fragment dword (lane_slot ls, dword v):
  //   K = (ls>>4)*16 + 2v (+1 in high half), N(=kh) = ls&15 (cols 7..15 zero-padded)
  unsigned int* bhid = (unsigned int*)(smem + B_HI_OFF);
  unsigned int* blod = (unsigned int*)(smem + B_LO_OFF);
  for (int i = tid; i < NSTEP * 256; i += 128) {
    int s   = i >> 8;
    int rem = i & 255;
    int ls  = rem >> 3;
    int v   = rem & 7;
    int kw  = s >> 3;
    int c0  = (s & 7) * 32;
    int n   = ls & 15;
    int ke  = (ls >> 4) * 16 + 2 * v;
    int ca  = c0 + ke, cb = ca + 1;
    float wa = (n < KS) ? w[ca * (KS * KS) + n * KS + kw] : 0.f;
    float wb = (n < KS) ? w[cb * (KS * KS) + n * KS + kw] : 0.f;
    unsigned short wah = bf16_rn(wa), wbh = bf16_rn(wb);
    unsigned short wal = bf16_rn(wa - bf16_tof(wah));
    unsigned short wbl = bf16_rn(wb - bf16_tof(wbh));
    int idx = s * 256 + ls * 8 + v;
    bhid[idx] = (unsigned)wah | ((unsigned)wbh << 16);
    blod[idx] = (unsigned)wal | ((unsigned)wbl << 16);
  }

  // zero the per-wave output accumulators
  float* oacc = (float*)(smem + OUT_OFF);
  for (int i = tid; i < NWAVE * TILE_H * TILE_M; i += 128) oacc[i] = 0.f;

  unsigned short* xhi = (unsigned short*)(smem + XT_HI_OFF);
  unsigned short* xlo = (unsigned short*)(smem + XT_LO_OFF);

  for (int hh = 0; hh < TILE_H + KS - 1; ++hh) {
    __syncthreads();                    // previous compute done reading XT

    // ---- stage input row hh: x[b, :, h, ow0..ow0+69] transposed -> XT[col][c], bf16 hi/lo
    // While staging row hh, prefetch row hh+1 into cache (global_prefetch_b8).
    int hrow = oh0 + hh;     if (hrow > IMG - 1) hrow = IMG - 1;
    int hnxt = oh0 + hh + 1; if (hnxt > IMG - 1) hnxt = IMG - 1;
    for (unsigned i = tid; i < POSW * IN_C; i += 128) {
      unsigned c   = i / POSW;
      unsigned col = i - c * POSW;
      int gcol = ow0 + (int)col; if (gcol > IMG - 1) gcol = IMG - 1;
      const float* gp = &x[(((size_t)b * IN_C + c) * IMG + (size_t)hrow) * IMG + gcol];
      if ((col & 31u) == 0u && hh + 1 < TILE_H + KS - 1) {
        const float* np = &x[(((size_t)b * IN_C + c) * IMG + (size_t)hnxt) * IMG + gcol];
        __builtin_prefetch(np, 0, 1);   // -> global_prefetch_b8 (next row, ~per cacheline)
      }
      float val = *gp;
      unsigned short h16 = bf16_rn(val);
      unsigned short l16 = bf16_rn(val - bf16_tof(h16));
      unsigned d = col * IN_C + c;
      xhi[d] = h16;
      xlo[d] = l16;
    }
    __syncthreads();

    // ---- per wave: z[m=ow, n=kh] = sum_{c,kw} x * w  (3-term bf16 split, f32 acc)
    // Three independent accumulator chains so the 3 WMMAs per K-step have no
    // D->C dependency on each other (better XDL pipelining); merged at scatter.
    v8f acc0 = {0.f, 0.f, 0.f, 0.f, 0.f, 0.f, 0.f, 0.f};
    v8f acc1 = acc0;
    v8f acc2 = acc0;
    const int m     = lane & 15;
    const int chalf = (lane >> 4) * 16;
    for (int s = 0; s < NSTEP; ++s) {
      const int kw  = s >> 3;
      const int c0  = (s & 7) * 32;
      const int pos = wv * TILE_M + m + kw;
      const uint4* pah = (const uint4*)(smem + XT_HI_OFF + ((size_t)(pos * IN_C + c0 + chalf) * 2));
      const uint4* pal = (const uint4*)(smem + XT_LO_OFF + ((size_t)(pos * IN_C + c0 + chalf) * 2));
      const uint4* pbh = (const uint4*)(smem + B_HI_OFF + s * 1024 + lane * 32);
      const uint4* pbl = (const uint4*)(smem + B_LO_OFF + s * 1024 + lane * 32);
      Frag16 Ah, Al, Bh, Bl;
      Ah.q[0] = pah[0]; Ah.q[1] = pah[1];   // K 0..15 | 16..31 (contiguous per lane half)
      Al.q[0] = pal[0]; Al.q[1] = pal[1];
      Bh.q[0] = pbh[0]; Bh.q[1] = pbh[1];
      Bl.q[0] = pbl[0]; Bl.q[1] = pbl[1];
      acc0 = __builtin_amdgcn_wmma_f32_16x16x32_bf16(false, Ah.v, false, Bh.v, (short)0, acc0, false, false);
      acc1 = __builtin_amdgcn_wmma_f32_16x16x32_bf16(false, Ah.v, false, Bl.v, (short)0, acc1, false, false);
      acc2 = __builtin_amdgcn_wmma_f32_16x16x32_bf16(false, Al.v, false, Bh.v, (short)0, acc2, false, false);
    }

    // ---- scatter z into output-row accumulators: out[r=hh-kh][m] += z[m][kh]
    // D layout: lane L holds M = v + 8*(L>>4) (=ow), N = L&15 (=kh); addresses are
    // disjoint across lanes for a fixed hh -> ds_add_f32 per element, race-free.
    const int kh = lane & 15;
    const int r  = hh - kh;
    if (kh < KS && r >= 0 && r < TILE_H) {
      float* op = oacc + wv * (TILE_H * TILE_M) + r * TILE_M + ((lane >> 4) * 8);
      atomicAdd(op + 0, acc0[0] + acc1[0] + acc2[0]);
      atomicAdd(op + 1, acc0[1] + acc1[1] + acc2[1]);
      atomicAdd(op + 2, acc0[2] + acc1[2] + acc2[2]);
      atomicAdd(op + 3, acc0[3] + acc1[3] + acc2[3]);
      atomicAdd(op + 4, acc0[4] + acc1[4] + acc2[4]);
      atomicAdd(op + 5, acc0[5] + acc1[5] + acc2[5]);
      atomicAdd(op + 6, acc0[6] + acc1[6] + acc2[6]);
      atomicAdd(op + 7, acc0[7] + acc1[7] + acc2[7]);
    }
  }
  __syncthreads();

  // ---- store tile (+bias), masking the ragged right/bottom edges
  const float bv = bias[0];
  for (int i = tid; i < NWAVE * TILE_H * TILE_M; i += 128) {
    int wvi = i >> 8;
    int r   = (i >> 4) & 15;
    int mm  = i & 15;
    int oh  = oh0 + r;
    int ow  = ow0 + wvi * TILE_M + mm;
    if (oh < OUTW && ow < OUTW)
      out[((size_t)b * OUTW + oh) * OUTW + ow] = oacc[i] + bv;
  }
}

extern "C" void kernel_launch(void* const* d_in, const int* in_sizes, int n_in,
                              void* d_out, int out_size, void* d_ws, size_t ws_size,
                              hipStream_t stream) {
  (void)in_sizes; (void)n_in; (void)d_ws; (void)ws_size; (void)out_size;
  const float* x    = (const float*)d_in[0];
  const float* w    = (const float*)d_in[1];
  const float* bias = (const float*)d_in[2];
  float* out        = (float*)d_out;
  dim3 grid(16, (OUTW + TILE_H - 1) / TILE_H, (OUTW + TILE_W - 1) / TILE_W); // 16 x 14 x 4
  conv7x7_wmma_bf16x3<<<grid, dim3(128, 1, 1), 0, stream>>>(x, w, bias, out);
}
